// BiLSTMClassifier_87746181857705
// MI455X (gfx1250) — matrix operand order repaired and checked by reading, not compiled, not run on hardware
//
#include <hip/hip_runtime.h>
#include <hip/hip_bf16.h>
#include <cstddef>

// Problem sizes (fixed by the reference)
#define BB    128
#define TT    256
#define DD    512
#define HH    256
#define CCC   96
#define FOURH 1024
#define FOURC 384

typedef __attribute__((ext_vector_type(16))) _Float16 v16h;
typedef __attribute__((ext_vector_type(8)))  _Float16 v8h;
typedef __attribute__((ext_vector_type(8)))  float    v8f;

__device__ __forceinline__ float sigmoidf_(float x){ return 1.0f/(1.0f+__expf(-x)); }

__device__ __forceinline__ v16h cat8_(v8h lo, v8h hi){
  return __builtin_shufflevector(lo, hi, 0,1,2,3,4,5,6,7,8,9,10,11,12,13,14,15);
}

// ---------------- conversion / fill ----------------
__global__ void f32_to_f16_kernel(const float* __restrict__ src, _Float16* __restrict__ dst, size_t n){
  size_t i = (size_t)blockIdx.x*blockDim.x + threadIdx.x;
  if (i < n) dst[i] = (_Float16)src[i];
}
// W[K,N] fp32 -> WT[N,K] f16 (weights become contiguous-K B-operands)
__global__ void f32_to_f16_T_kernel(const float* __restrict__ src, _Float16* __restrict__ dst,
                                    int K, int N){
  int idx = blockIdx.x*blockDim.x + threadIdx.x;
  if (idx < K*N){
    int k = idx / N, n = idx - k*N;
    dst[(size_t)n*K + k] = (_Float16)src[idx];
  }
}
__global__ void fill_f32_kernel(float* __restrict__ p, float v, int n){
  int i = blockIdx.x*blockDim.x + threadIdx.x;
  if (i < n) p[i] = v;
}
__global__ void fill_f16_kernel(_Float16* __restrict__ p, int n){
  int i = blockIdx.x*blockDim.x + threadIdx.x;
  if (i < n) p[i] = (_Float16)0.0f;
}

// ---------------- big input-projection GEMM ----------------
// C[M,N] = A[M,K] * B[K,N] + bias[N]; A row-major f16, B given TRANSPOSED: BT[N,K] f16.
// grid = (N/128, M/16), block = (32,4): each wave computes a 16x32 strip (2 tiles, shared A frag).
__global__ void gemm_f16_wmma(const _Float16* __restrict__ A, const _Float16* __restrict__ BT,
                              const float* __restrict__ bias, float* __restrict__ Cout,
                              int N, int K){
  const int lane = threadIdx.x;
  const int half = lane >> 4;         // K-half selector (ISA §7.12.2 layouts)
  const int r    = lane & 15;
  const int n0   = (blockIdx.x * blockDim.y + threadIdx.y) * 32;
  const int m0   = blockIdx.y * 16;
  v8f acc0 = {0.f,0.f,0.f,0.f,0.f,0.f,0.f,0.f};
  v8f acc1 = {0.f,0.f,0.f,0.f,0.f,0.f,0.f,0.f};
  const _Float16* Arow  = A  + (size_t)(m0 + r) * K;
  const _Float16* Brow0 = BT + (size_t)(n0 + r) * K;        // column n0+r of B
  const _Float16* Brow1 = BT + (size_t)(n0 + 16 + r) * K;   // column n0+16+r of B
  for (int k0 = 0; k0 < K; k0 += 32){
    __builtin_prefetch(Arow + k0 + 128, 0, 3);
    // A 16x32 frag: two contiguous 8-elem chunks at k0+8*half and k0+16+8*half
    v16h afrag = cat8_(*(const v8h*)(Arow + k0 + 8*half),
                       *(const v8h*)(Arow + k0 + 16 + 8*half));
    // B 32x16 frag: two contiguous 8-elem chunks of BT row at k0+8*half and k0+16+8*half
    v16h bfrag0 = cat8_(*(const v8h*)(Brow0 + k0 + 8*half),
                        *(const v8h*)(Brow0 + k0 + 16 + 8*half));
    v16h bfrag1 = cat8_(*(const v8h*)(Brow1 + k0 + 8*half),
                        *(const v8h*)(Brow1 + k0 + 16 + 8*half));
    acc0 = __builtin_amdgcn_wmma_f32_16x16x32_f16(false, afrag, false, bfrag0,
                                                  (short)0, acc0, false, false);
    acc1 = __builtin_amdgcn_wmma_f32_16x16x32_f16(false, afrag, false, bfrag1,
                                                  (short)0, acc1, false, false);
  }
  const float bv0 = bias[n0 + r];
  const float bv1 = bias[n0 + 16 + r];
  #pragma unroll
  for (int i = 0; i < 8; ++i){        // D: M = i + 8*half, N = lane&15
    int m = m0 + i + 8*half;
    Cout[(size_t)m*N + n0 + r]      = acc0[i] + bv0;
    Cout[(size_t)m*N + n0 + 16 + r] = acc1[i] + bv1;
  }
}

// ---------------- recurrent step GEMM ----------------
// Z[B,N] = Hp[B,K] * Wh[K,N] + xg[:, t, :]; Wh given transposed WhT[N,K].
// grid = (N/128, BB/16), block = (32,4)
__global__ void recur_gemm_wmma(const _Float16* __restrict__ Hp, const _Float16* __restrict__ WhT,
                                const float* __restrict__ xg, float* __restrict__ Z,
                                int t, int N, int K){
  const int lane = threadIdx.x;
  const int half = lane >> 4;
  const int r    = lane & 15;
  const int n0   = (blockIdx.x * blockDim.y + threadIdx.y) * 32;
  const int m0   = blockIdx.y * 16;
  v8f acc0 = {0.f,0.f,0.f,0.f,0.f,0.f,0.f,0.f};
  v8f acc1 = {0.f,0.f,0.f,0.f,0.f,0.f,0.f,0.f};
  const _Float16* Arow  = Hp  + (size_t)(m0 + r) * K;
  const _Float16* Brow0 = WhT + (size_t)(n0 + r) * K;
  const _Float16* Brow1 = WhT + (size_t)(n0 + 16 + r) * K;
  for (int k0 = 0; k0 < K; k0 += 32){
    v16h afrag  = cat8_(*(const v8h*)(Arow + k0 + 8*half),
                        *(const v8h*)(Arow + k0 + 16 + 8*half));
    v16h bfrag0 = cat8_(*(const v8h*)(Brow0 + k0 + 8*half),
                        *(const v8h*)(Brow0 + k0 + 16 + 8*half));
    v16h bfrag1 = cat8_(*(const v8h*)(Brow1 + k0 + 8*half),
                        *(const v8h*)(Brow1 + k0 + 16 + 8*half));
    acc0 = __builtin_amdgcn_wmma_f32_16x16x32_f16(false, afrag, false, bfrag0,
                                                  (short)0, acc0, false, false);
    acc1 = __builtin_amdgcn_wmma_f32_16x16x32_f16(false, afrag, false, bfrag1,
                                                  (short)0, acc1, false, false);
  }
  #pragma unroll
  for (int i = 0; i < 8; ++i){
    int m = m0 + i + 8*half;                       // batch row
    const float* xr = xg + ((size_t)m*TT + t)*N;
    Z[(size_t)m*N + n0 + r]      = acc0[i] + xr[n0 + r];
    Z[(size_t)m*N + n0 + 16 + r] = acc1[i] + xr[n0 + 16 + r];
  }
}

// ---------------- LSTM pointwise (tanh cell act) ----------------
__global__ void lstm_pointwise_tanh(const float* __restrict__ Z, float* __restrict__ c,
                                    _Float16* __restrict__ h16, _Float16* __restrict__ hdst,
                                    int ld, int hoff, int t){
  int idx = blockIdx.x*blockDim.x + threadIdx.x;   // over B*H
  int b = idx / HH, j = idx - b*HH;
  const float* zr = Z + (size_t)b*FOURH;
  float iv = sigmoidf_(zr[j]);
  float fv = sigmoidf_(zr[HH  + j]);
  float gv = tanhf   (zr[2*HH + j]);
  float ov = sigmoidf_(zr[3*HH + j]);
  float cj = fv * c[idx] + iv * gv;
  c[idx] = cj;
  float h = ov * tanhf(cj);
  h16[idx] = (_Float16)h;
  hdst[((size_t)b*TT + t)*ld + hoff + j] = (_Float16)h;
}

// ---------------- CRF-LSTM pointwise (softmax cell act over C) ----------------
__global__ void crf_pointwise(const float* __restrict__ Z, float* __restrict__ c,
                              _Float16* __restrict__ h16, float* __restrict__ pot,
                              int t, int accumulate){
  int b = blockIdx.x;
  int j = threadIdx.x;                    // 0..127 (first 96 active for data)
  __shared__ float red[128];
  const float* zr = Z + (size_t)b*FOURC;
  float iv=0.f, fv=0.f, ov=0.f, gg=0.f;
  if (j < CCC){
    iv = zr[j]; fv = zr[CCC+j]; gg = zr[2*CCC+j]; ov = zr[3*CCC+j];
  }
  // softmax(g)
  red[j] = (j < CCC) ? gg : -1e30f;  __syncthreads();
  for (int s = 64; s > 0; s >>= 1){ if (j < s) red[j] = fmaxf(red[j], red[j+s]); __syncthreads(); }
  float gmax = red[0]; __syncthreads();
  float ge = (j < CCC) ? __expf(gg - gmax) : 0.f;
  red[j] = ge; __syncthreads();
  for (int s = 64; s > 0; s >>= 1){ if (j < s) red[j] += red[j+s]; __syncthreads(); }
  float gsum = red[0]; __syncthreads();
  float gact = ge / gsum;
  // cell update
  float cj = 0.f;
  if (j < CCC){
    float si = sigmoidf_(iv), sf = sigmoidf_(fv);
    cj = sf * c[b*CCC + j] + si * gact;
    c[b*CCC + j] = cj;
  }
  // softmax(c)
  red[j] = (j < CCC) ? cj : -1e30f; __syncthreads();
  for (int s = 64; s > 0; s >>= 1){ if (j < s) red[j] = fmaxf(red[j], red[j+s]); __syncthreads(); }
  float cmax = red[0]; __syncthreads();
  float ce = (j < CCC) ? __expf(cj - cmax) : 0.f;
  red[j] = ce; __syncthreads();
  for (int s = 64; s > 0; s >>= 1){ if (j < s) red[j] += red[j+s]; __syncthreads(); }
  float csum = red[0];
  if (j < CCC){
    float h = sigmoidf_(ov) * (ce / csum);
    h16[b*CCC + j] = (_Float16)h;
    size_t po = ((size_t)b*TT + t)*CCC + j;
    if (accumulate) pot[po] += h; else pot[po] = h;
  }
}

// ---------------- Viterbi: one workgroup per batch row, trans staged in LDS ----------------
__global__ void viterbi_kernel(const float* __restrict__ pot, const float* __restrict__ trans,
                               unsigned char* __restrict__ bp, int* __restrict__ decoded){
  __shared__ float s_trans[CCC*CCC];     // 36 KB of the 320 KB WGP LDS
  __shared__ float alpha[CCC];
  __shared__ float nalpha[CCC];
  int b = blockIdx.x;
  for (int i = threadIdx.x; i < CCC*CCC; i += blockDim.x) s_trans[i] = trans[i];
  int j = threadIdx.x;
  if (j < CCC) alpha[j] = pot[((size_t)b*TT)*CCC + j];
  __syncthreads();
  for (int t = 1; t < TT; ++t){
    if (j < CCC){
      float best = -1e30f; int bi = 0;
      #pragma unroll 4
      for (int i = 0; i < CCC; ++i){
        float v = alpha[i] + s_trans[i*CCC + j];
        if (v > best){ best = v; bi = i; }      // first-max, matches jnp.argmax
      }
      nalpha[j] = best + pot[((size_t)b*TT + t)*CCC + j];
      bp[((size_t)b*TT + t)*CCC + j] = (unsigned char)bi;
    }
    __syncthreads();
    if (j < CCC) alpha[j] = nalpha[j];
    __syncthreads();
  }
  if (j == 0){
    float best = -1e30f; int cur = 0;
    for (int i = 0; i < CCC; ++i){ if (alpha[i] > best){ best = alpha[i]; cur = i; } }
    decoded[(size_t)b*TT + TT - 1] = cur;
    for (int t = TT - 2; t >= 0; --t){
      cur = bp[((size_t)b*TT + t + 1)*CCC + cur];
      decoded[(size_t)b*TT + t] = cur;
    }
  }
}

// ======================= host orchestration =======================
extern "C" void kernel_launch(void* const* d_in, const int* in_sizes, int n_in,
                              void* d_out, int out_size, void* d_ws, size_t ws_size,
                              hipStream_t stream){
  (void)in_sizes; (void)n_in; (void)out_size; (void)ws_size;
  const float* x     = (const float*)d_in[0];
  const float* Wi1f  = (const float*)d_in[1];
  const float* Wh1f  = (const float*)d_in[2];
  const float* b1f   = (const float*)d_in[3];
  const float* Wi2f  = (const float*)d_in[4];
  const float* Wh2f  = (const float*)d_in[5];
  const float* b2f   = (const float*)d_in[6];
  const float* Wi1b  = (const float*)d_in[7];
  const float* Wh1b  = (const float*)d_in[8];
  const float* b1b   = (const float*)d_in[9];
  const float* Wi2b  = (const float*)d_in[10];
  const float* Wh2b  = (const float*)d_in[11];
  const float* b2b   = (const float*)d_in[12];
  const float* Wif   = (const float*)d_in[13];
  const float* Whf   = (const float*)d_in[14];
  const float* bf_   = (const float*)d_in[15];
  const float* Wib   = (const float*)d_in[16];
  const float* Whb   = (const float*)d_in[17];
  const float* bb_   = (const float*)d_in[18];
  const float* trans = (const float*)d_in[19];

  int*   decoded = (int*)d_out;                       // [B,T] int32 bits
  float* pot     = (float*)d_out + (size_t)BB*TT;     // [B,T,C] fp32

  // workspace carve-out
  char* wsp = (char*)d_ws;
  size_t off = 0;
  auto carve = [&](size_t bytes)->char*{
    char* p = wsp + off; off += (bytes + 255) & ~(size_t)255; return p;
  };
  _Float16* x16     = (_Float16*)carve((size_t)BB*TT*DD*2);
  _Float16* Wi1fT   = (_Float16*)carve((size_t)DD*FOURH*2);     // [4H, D]
  _Float16* Wh1fT   = (_Float16*)carve((size_t)HH*FOURH*2);     // [4H, H]
  _Float16* Wi2fT   = (_Float16*)carve((size_t)HH*FOURH*2);
  _Float16* Wh2fT   = (_Float16*)carve((size_t)HH*FOURH*2);
  _Float16* Wi1bT   = (_Float16*)carve((size_t)DD*FOURH*2);
  _Float16* Wh1bT   = (_Float16*)carve((size_t)HH*FOURH*2);
  _Float16* Wi2bT   = (_Float16*)carve((size_t)HH*FOURH*2);
  _Float16* Wh2bT   = (_Float16*)carve((size_t)HH*FOURH*2);
  _Float16* WifT    = (_Float16*)carve((size_t)2*HH*FOURC*2);   // [4C, 2H]
  _Float16* WhfT    = (_Float16*)carve((size_t)CCC*FOURC*2);    // [4C, C]
  _Float16* WibT    = (_Float16*)carve((size_t)2*HH*FOURC*2);
  _Float16* WhbT    = (_Float16*)carve((size_t)CCC*FOURC*2);
  float*    xgbuf   = (float*)   carve((size_t)BB*TT*FOURH*4);  // reused by all 6 LSTMs
  _Float16* hseq16  = (_Float16*)carve((size_t)BB*TT*HH*2);     // layer-1 output (per direction)
  _Float16* hcat16  = (_Float16*)carve((size_t)BB*TT*2*HH*2);   // [hf | hb]
  float*    zbuf    = (float*)   carve((size_t)BB*FOURH*4);
  float*    cstate  = (float*)   carve((size_t)BB*HH*4);
  _Float16* hstate  = (_Float16*)carve((size_t)BB*HH*2);
  unsigned char* bpb = (unsigned char*)carve((size_t)BB*TT*CCC);

  f32_to_f16_kernel<<<(unsigned)(((size_t)BB*TT*DD + 255)/256), 256, 0, stream>>>(
      x, x16, (size_t)BB*TT*DD);
  auto convT = [&](const float* s, _Float16* d, int K, int N){
    f32_to_f16_T_kernel<<<(unsigned)((K*N + 255)/256), 256, 0, stream>>>(s, d, K, N);
  };
  convT(Wi1f, Wi1fT, DD,  FOURH);  convT(Wh1f, Wh1fT, HH,  FOURH);
  convT(Wi2f, Wi2fT, HH,  FOURH);  convT(Wh2f, Wh2fT, HH,  FOURH);
  convT(Wi1b, Wi1bT, DD,  FOURH);  convT(Wh1b, Wh1bT, HH,  FOURH);
  convT(Wi2b, Wi2bT, HH,  FOURH);  convT(Wh2b, Wh2bT, HH,  FOURH);
  convT(Wif,  WifT,  2*HH, FOURC); convT(Whf,  WhfT,  CCC, FOURC);
  convT(Wib,  WibT,  2*HH, FOURC); convT(Whb,  WhbT,  CCC, FOURC);

  auto big_gemm = [&](const _Float16* A, const _Float16* BT, const float* bias, int N, int K){
    gemm_f16_wmma<<<dim3(N/128, (BB*TT)/16), dim3(32,4), 0, stream>>>(A, BT, bias, xgbuf, N, K);
  };
  auto run_h_lstm = [&](const _Float16* WhT, _Float16* hdst, int ld, int hoff, bool rev){
    fill_f16_kernel<<<(BB*HH+255)/256, 256, 0, stream>>>(hstate, BB*HH);
    fill_f32_kernel<<<(BB*HH+255)/256, 256, 0, stream>>>(cstate, 0.f, BB*HH);
    for (int s = 0; s < TT; ++s){
      int t = rev ? (TT - 1 - s) : s;
      recur_gemm_wmma<<<dim3(FOURH/128, BB/16), dim3(32,4), 0, stream>>>(
          hstate, WhT, xgbuf, zbuf, t, FOURH, HH);
      lstm_pointwise_tanh<<<(BB*HH)/256, 256, 0, stream>>>(
          zbuf, cstate, hstate, hdst, ld, hoff, t);
    }
  };
  auto run_c_lstm = [&](const _Float16* WhT, bool rev, int accumulate){
    fill_f16_kernel<<<(BB*CCC+255)/256, 256, 0, stream>>>(hstate, BB*CCC);
    fill_f32_kernel<<<(BB*CCC+255)/256, 256, 0, stream>>>(cstate, 0.f, BB*CCC);
    for (int s = 0; s < TT; ++s){
      int t = rev ? (TT - 1 - s) : s;
      recur_gemm_wmma<<<dim3(FOURC/128, BB/16), dim3(32,4), 0, stream>>>(
          hstate, WhT, xgbuf, zbuf, t, FOURC, CCC);
      crf_pointwise<<<BB, 128, 0, stream>>>(zbuf, cstate, hstate, pot, t, accumulate);
    }
  };

  // forward stack: layer1f -> layer2f (into hcat[:, :, 0:256])
  big_gemm(x16,    Wi1fT, b1f, FOURH, DD);
  run_h_lstm(Wh1fT, hseq16, HH, 0, false);
  big_gemm(hseq16, Wi2fT, b2f, FOURH, HH);
  run_h_lstm(Wh2fT, hcat16, 2*HH, 0, false);

  // backward stack: layer1b -> layer2b (into hcat[:, :, 256:512])
  big_gemm(x16,    Wi1bT, b1b, FOURH, DD);
  run_h_lstm(Wh1bT, hseq16, HH, 0, true);
  big_gemm(hseq16, Wi2bT, b2b, FOURH, HH);
  run_h_lstm(Wh2bT, hcat16, 2*HH, HH, true);

  // CRF potentials: pf (forward, assign) + pb (backward, accumulate)
  big_gemm(hcat16, WifT, bf_, FOURC, 2*HH);
  run_c_lstm(WhfT, false, 0);
  big_gemm(hcat16, WibT, bb_, FOURC, 2*HH);
  run_c_lstm(WhbT, true, 1);

  // Viterbi decode
  viterbi_kernel<<<BB, 128, 0, stream>>>(pot, trans, bpb, decoded);
}
